// SpikeModel_47811575939365
// MI455X (gfx1250) — compile-verified, matrix operand order checked by reading, not run on hardware
//
#include <hip/hip_runtime.h>
#include <hip/hip_bf16.h>
#include <stddef.h>

typedef __attribute__((ext_vector_type(16))) _Float16 v16h;
typedef __attribute__((ext_vector_type(8)))  float    v8f;

#define TSTEPS 20
#define DECAY  0.2f
#define THRESH 0.5f

// ---------------- workspace layout (bytes) ----------------
// Weights stored pre-permuted into WMMA A-fragment order:
//   [mtile(2)][kc(KPAD/32)][lane(32)][j(16)] f16, 32B contiguous per (mtile,kc,lane)
// wq1: 2*1*32*16 f16  = 2048 B              @ 0
// wq2: 2*9*32*16 f16  = 18432 B             @ 2048
// wq3: 2*9*32*16 f16  = 18432 B             @ 20480
// spk1: 32*32*64*64 f16                     @ 40960    (8 MB)
// spk2: 32*32*32*32 f16                     @ 8429568  (2 MB)
// spk3: 32*32*16*16 f16                     @ 10526720 (0.5 MB)
// pot1: f32                                 @ 11051008 (16 MB)
// pot2: f32                                 @ 27828224 (4 MB)
// pot3: f32                                 @ 32022528 (1 MB)  total ~31.6 MB
#define OFF_WQ1  ((size_t)0)
#define OFF_WQ2  ((size_t)2048)
#define OFF_WQ3  ((size_t)20480)
#define OFF_SPK1 ((size_t)40960)
#define OFF_SPK2 ((size_t)8429568)
#define OFF_SPK3 ((size_t)10526720)
#define OFF_POT1 ((size_t)11051008)
#define OFF_POT2 ((size_t)27828224)
#define OFF_POT3 ((size_t)32022528)

// Pre-permute one layer's fp32 weights into f16 WMMA A-fragment order.
// Output element i: j=i&15, lane=(i>>4)&31, kc=(i>>9)%NCH, mtile=(i>>9)/NCH.
// ISA K-striping: kin = (j&7) + (j&8 ? 16:0) + (lane&16 ? 8:0);  k = kc*32+kin.
__device__ __forceinline__ void permute_one(const float* __restrict__ w,
                                            _Float16* __restrict__ q,
                                            int i, int NCH, int KTOT) {
    int j    = i & 15;
    int lane = (i >> 4) & 31;
    int g    = i >> 9;              // mtile*NCH + kc
    int mtile = g / NCH;
    int kc    = g - mtile * NCH;
    int kin = (j & 7) + ((j & 8) << 1) + ((lane & 16) >> 1);
    int k   = kc * 32 + kin;
    int row = mtile * 16 + (lane & 15);
    q[i] = (k < KTOT) ? (_Float16)w[row * KTOT + k] : (_Float16)0.f;
}

__global__ void convert_weights(const float* __restrict__ w1,
                                const float* __restrict__ w2,
                                const float* __restrict__ w3,
                                _Float16* __restrict__ q1,
                                _Float16* __restrict__ q2,
                                _Float16* __restrict__ q3) {
    int i = blockIdx.x * blockDim.x + threadIdx.x;
    if (i < 1024) {
        permute_one(w1, q1, i, 1, 18);          // layer 1: K=18 padded to 32
    } else if (i < 1024 + 9216) {
        permute_one(w2, q2, i - 1024, 9, 288);  // layer 2
    } else if (i < 1024 + 18432) {
        permute_one(w3, q3, i - 1024 - 9216, 9, 288);  // layer 3
    }
}

// One LIF-conv layer at one timestep.
// Block = 64 threads = 2 waves; wave m computes cout [16m,16m+16) for a 1x16
// output strip (row oh, cols ow0..ow0+15). The block stages the strip's im2col
// matrix (KPAD x 16 f16) into LDS *already permuted into WMMA B-fragment
// order*, so each lane's B fragment is a single contiguous 32B LDS read.
template <int CIN, int HIN, int WIN, int HOUT, int WOUT, int KTOT, int KPAD,
          bool FIRST_LAYER, bool LAST_LAYER>
__global__ __launch_bounds__(64) void lif_conv_wmma(
    const float* __restrict__ inF,     // layer 1: raw input, innermost stride T
    const _Float16* __restrict__ inS,  // layers 2/3: prev layer spikes (f16)
    const _Float16* __restrict__ wq,   // fragment-ordered f16 weights
    const float* __restrict__ bias,    // [32]
    float* __restrict__ pot,           // membrane state [img][32][HOUT][WOUT]
    _Float16* __restrict__ spk,        // this layer's spikes (read prev-t, write new)
    float* __restrict__ out,           // last layer: (bs, 8192, T)
    int t, int first) {
    constexpr int TILES_X = WOUT / 16;
    constexpr int NCH = KPAD / 32;
    const int tile = blockIdx.x;
    const int img  = blockIdx.y;
    const int oh   = tile / TILES_X;
    const int ow0  = (tile % TILES_X) * 16;

    const int tid   = threadIdx.x;  // 0..63
    const int lane  = tid & 31;
    const int waveM = tid >> 5;     // M-tile: cout 0-15 or 16-31
    const int col   = lane & 15;
    const bool hiHalf = (lane >= 16);

    __shared__ __attribute__((aligned(32))) _Float16 sB[KPAD * 16];

    // ---- stage im2col in B-fragment order ----
    // element e: col=e&15, k=e>>4; value = in[cin][2*oh-1+kh][2*(ow0+col)-1+kw]
    // scatter to sB[(kc*32 + fragLane)*16 + j]
    const int ih0 = 2 * oh - 1;
    const int iw0 = 2 * ow0 - 1;
    for (int e = tid; e < KPAD * 16; e += 64) {
        int c = e & 15;
        int k = e >> 4;
        _Float16 v = (_Float16)0.f;
        if (k < KTOT) {
            int cin = k / 9;
            int r9  = k - cin * 9;
            int kh  = r9 / 3;
            int kw  = r9 - kh * 3;
            int ih = ih0 + kh;
            int iw = iw0 + 2 * c + kw;
            if (ih >= 0 && ih < HIN && iw >= 0 && iw < WIN) {
                if (FIRST_LAYER) {
                    v = (_Float16)inF[((((size_t)img * CIN + cin) * HIN + ih) * WIN + iw) * TSTEPS + t];
                } else {
                    v = inS[(((size_t)img * CIN + cin) * HIN + ih) * WIN + iw];
                }
            }
        }
        int kc  = k >> 5;
        int kin = k & 31;
        int j2       = (kin & 7) | ((kin & 16) >> 1);   // fragment element
        int fragLane = c | ((kin & 8) << 1);            // fragment lane
        sB[(kc * 32 + fragLane) * 16 + j2] = v;
    }
    __syncthreads();

    // ---- implicit GEMM: D(16x16) = W(16xK) * im2col(Kx16) + bias ----
    const _Float16* wbase = wq + (size_t)waveM * NCH * 32 * 16;
    const int coutBase = waveM * 16 + (hiHalf ? 8 : 0);  // C/D: M = r + (hi?8:0)

    v8f acc;
#pragma unroll
    for (int r = 0; r < 8; ++r) acc[r] = bias[coutBase + r];

#pragma unroll
    for (int kc = 0; kc < NCH; ++kc) {
        v16h a = *(const v16h*)(wbase + (size_t)(kc * 32 + lane) * 16);
        v16h b = *(const v16h*)(&sB[(kc * 32 + lane) * 16]);
        acc = __builtin_amdgcn_wmma_f32_16x16x32_f16(
            /*neg_a=*/false, a, /*neg_b=*/false, b,
            /*c_mod=*/(short)0, acc, /*reuse_a=*/false, /*reuse_b=*/false);
    }

    // ---- LIF update: u = pot*DECAY*(1-spk_prev) + conv; spike = u > THRESH ----
#pragma unroll
    for (int r = 0; r < 8; ++r) {
        int cout = coutBase + r;
        int ow   = ow0 + col;
        size_t p = (((size_t)img * 32 + cout) * HOUT + oh) * WOUT + ow;
        float u = acc[r];
        if (!first) {
            float pprev = pot[p];
            float sprev = (float)spk[p];
            u += pprev * DECAY * (1.0f - sprev);
        }
        float sp = (u > THRESH) ? 1.0f : 0.0f;
        pot[p] = u;
        spk[p] = (_Float16)sp;
        if (LAST_LAYER) {
            size_t oidx = ((size_t)img * 8192 + ((size_t)cout * 256 + oh * 16 + ow)) * TSTEPS + t;
            out[oidx] = sp;
        }
    }
}

extern "C" void kernel_launch(void* const* d_in, const int* in_sizes, int n_in,
                              void* d_out, int out_size, void* d_ws, size_t ws_size,
                              hipStream_t stream) {
    const float* input = (const float*)d_in[0];
    const float* w1 = (const float*)d_in[1];
    const float* b1 = (const float*)d_in[2];
    const float* w2 = (const float*)d_in[3];
    const float* b2 = (const float*)d_in[4];
    const float* w3 = (const float*)d_in[5];
    const float* b3 = (const float*)d_in[6];
    float* out = (float*)d_out;

    char* ws = (char*)d_ws;
    _Float16* wq1  = (_Float16*)(ws + OFF_WQ1);
    _Float16* wq2  = (_Float16*)(ws + OFF_WQ2);
    _Float16* wq3  = (_Float16*)(ws + OFF_WQ3);
    _Float16* spk1 = (_Float16*)(ws + OFF_SPK1);
    _Float16* spk2 = (_Float16*)(ws + OFF_SPK2);
    _Float16* spk3 = (_Float16*)(ws + OFF_SPK3);
    float* pot1 = (float*)(ws + OFF_POT1);
    float* pot2 = (float*)(ws + OFF_POT2);
    float* pot3 = (float*)(ws + OFF_POT3);

    // quantize + permute weights into WMMA A-fragment order (19456 elements)
    convert_weights<<<(19456 + 255) / 256, 256, 0, stream>>>(w1, w2, w3, wq1, wq2, wq3);

    for (int t = 0; t < TSTEPS; ++t) {
        int first = (t == 0) ? 1 : 0;
        // layer 1: 2ch 128x128 -> 32ch 64x64, K=18 (pad 32); 256 strips x 32 imgs
        lif_conv_wmma<2, 128, 128, 64, 64, 18, 32, true, false>
            <<<dim3(256, 32), 64, 0, stream>>>(input, nullptr, wq1, b1, pot1, spk1,
                                               nullptr, t, first);
        // layer 2: 32ch 64x64 -> 32ch 32x32, K=288; 64 strips x 32 imgs
        lif_conv_wmma<32, 64, 64, 32, 32, 288, 288, false, false>
            <<<dim3(64, 32), 64, 0, stream>>>(nullptr, spk1, wq2, b2, pot2, spk2,
                                              nullptr, t, first);
        // layer 3: 32ch 32x32 -> 32ch 16x16, K=288; 16 strips x 32 imgs; writes d_out
        lif_conv_wmma<32, 32, 32, 16, 16, 288, 288, false, true>
            <<<dim3(16, 32), 64, 0, stream>>>(nullptr, spk2, wq3, b3, pot3, spk3,
                                              out, t, first);
    }
}
